// FP4Linear_67027259621403
// MI455X (gfx1250) — matrix operand order repaired and checked by reading, not compiled, not run on hardware
//
#include <hip/hip_runtime.h>
#include <hip/hip_bf16.h>

typedef __attribute__((ext_vector_type(16))) __bf16       v16bf;
typedef __attribute__((ext_vector_type(8)))  __bf16       v8bf;
typedef __attribute__((ext_vector_type(8)))  float        v8f;
typedef __attribute__((ext_vector_type(4)))  int          v4i;
typedef __attribute__((ext_vector_type(8)))  int          v8i;
typedef __attribute__((ext_vector_type(4)))  unsigned int v4u;

#define S_DIM   2048
#define IN_DIM  4096
#define OUT_DIM 11008
#define QBLK    64

#define TM 128     // block M tile
#define TN 128     // block N tile
#define TK 64      // K step per iteration (== quant block size)
#define LDSW 72    // padded row stride in bf16 elems (144B = 32 DW + 4 DW pad)
#define NIT (IN_DIM / TK)   // 64 iterations

union FragBF { v16bf v; v8bf h[2]; };

__global__ __launch_bounds__(256)
void fp4_linear_wmma(const __bf16* __restrict__ x,
                     const int*    __restrict__ codes,
                     const float*  __restrict__ absmax,
                     const __bf16* __restrict__ bias,
                     __bf16*       __restrict__ out)
{
    __shared__ __bf16 xs[2][TM * LDSW];   // x tiles [128 x 64] (TDM, padded)
    __shared__ __bf16 ws[2][TN * LDSW];   // W tiles [128 x 64] (dequantized)
    __shared__ float  tab[16];            // FP4 codebook

    const int tid   = threadIdx.x;
    const int lane  = tid & 31;
    const int wave  = tid >> 5;
    const int wm    = wave >> 1;        // 0..3  (M sub-tile, 32 rows)
    const int wn    = wave & 1;         // 0..1  (N sub-tile, 64 cols)
    const int lrow  = lane & 15;
    const int khalf = lane >> 4;        // 0 or 1

    const int mblk = blockIdx.y * TM;
    const int nblk = blockIdx.x * TN;

    if (tid < 16) {
        const float t[8] = {0.0f, 0.0052083333f, 0.6666667f, 1.0f,
                            0.3333333f, 0.5f, 0.16666667f, 0.25f};
        float v = t[tid & 7];
        tab[tid] = (tid & 8) ? -v : v;
    }

    // ---- W dequant assignment: 2 threads per n-row (32-wide k sections) ----
    const int  dqrow   = tid & 127;               // n within tile
    const int  dqk     = (tid >> 7) * 32;         // 0 or 32
    const int  gn      = nblk + dqrow;            // global n
    const long coderow = (long)gn * IN_DIM + dqk;

    // ---- TDM descriptor group1 (static): 2B elems, 2D, LDS row padding ----
    // d0: data_size=1(2B)<<16 | pad_en<<20 | pad_interval=4(32DW)<<22 | pad_amount=3(4DW)<<25
    v8i g1;
    g1[0] = (int)((1u << 16) | (1u << 20) | (4u << 22) | (3u << 25));
    g1[1] = (int)((IN_DIM & 0xFFFFu) << 16);                   // tensor_dim0[15:0]
    g1[2] = (int)((IN_DIM >> 16) | ((S_DIM & 0xFFFFu) << 16)); // dim0[31:16]|dim1[15:0]
    g1[3] = (int)((S_DIM >> 16) | (TK << 16));                 // dim1[31:16]|tile_dim0=64
    g1[4] = (int)TM;                                           // tile_dim1=128
    g1[5] = (int)IN_DIM;                                       // dim0_stride[31:0]
    g1[6] = 0;
    g1[7] = 0;
    const v4i gz = {0, 0, 0, 0};
    const unsigned long long xrow_base =
        (unsigned long long)(uintptr_t)(const void*)(x + (long)mblk * IN_DIM);

    // ---- stage: TDM x-tile + dequantize W-tile for iteration it2 into buf ----
    auto stage = [&](int it2, int buf) {
        if (wave == 0) {
            const unsigned long long gaddr =
                xrow_base + (unsigned long long)(it2 * TK) * 2u;
            const unsigned lds_off =
                (unsigned)(unsigned long long)(uintptr_t)(void*)&xs[buf][0];
            v4u g0;
            g0[0] = 1u;                                     // count=1
            g0[1] = lds_off;                                // lds_addr (bytes)
            g0[2] = (unsigned)(gaddr & 0xFFFFFFFFu);        // global_addr[31:0]
            g0[3] = (unsigned)((gaddr >> 32) & 0x1FFFFFFu)  // global_addr[56:32]
                    | (2u << 30);                           // type=2 (image)
#if __has_builtin(__builtin_amdgcn_tensor_load_to_lds)
#if __clang_major__ >= 23
            const v8i gz8 = {0, 0, 0, 0, 0, 0, 0, 0};
            __builtin_amdgcn_tensor_load_to_lds(g0, g1, gz, gz, gz8, 0);
#else
            __builtin_amdgcn_tensor_load_to_lds(g0, g1, gz, gz, 0);
#endif
#endif
        }
        const int  k0 = it2 * TK;
        const float s = absmax[(long)gn * (IN_DIM / QBLK) + it2];
        const v4i* cp = (const v4i*)(codes + coderow + k0);
        #pragma unroll
        for (int g = 0; g < 4; ++g) {
            v4i c0 = cp[2 * g];
            v4i c1 = cp[2 * g + 1];
            v8bf wv;
            wv[0] = (__bf16)(tab[c0.x] * s);
            wv[1] = (__bf16)(tab[c0.y] * s);
            wv[2] = (__bf16)(tab[c0.z] * s);
            wv[3] = (__bf16)(tab[c0.w] * s);
            wv[4] = (__bf16)(tab[c1.x] * s);
            wv[5] = (__bf16)(tab[c1.y] * s);
            wv[6] = (__bf16)(tab[c1.z] * s);
            wv[7] = (__bf16)(tab[c1.w] * s);
            *(v8bf*)&ws[buf][dqrow * LDSW + dqk + g * 8] = wv;
        }
        if (it2 + 1 < NIT)   // warm L2 for the next code tile
            __builtin_prefetch((const void*)(codes + coderow + k0 + TK), 0, 0);
    };

    v8f acc[2][4];
    #pragma unroll
    for (int i = 0; i < 2; ++i)
        #pragma unroll
        for (int j = 0; j < 4; ++j)
            acc[i][j] = {};

    const int arow0 = wm * 32 + lrow;        // A frag rows (mf = 0 / 1)
    const int arow1 = arow0 + 16;

    stage(0, 0);                              // preamble fill of buffer 0

    for (int it = 0; it < NIT; ++it) {
        const int cur = it & 1;

        if (wave == 0)                        // drain TDM before publishing tile
            __builtin_amdgcn_s_wait_tensorcnt(0);
        __syncthreads();                      // tile[cur] visible to all waves

        if (it + 1 < NIT)                     // overlap next stage with compute
            stage(it + 1, cur ^ 1);

        const __bf16* xb = &xs[cur][0];
        const __bf16* wb = &ws[cur][0];

        // ---- A fragments: 2 M-frags x 2 K-sections ----
        FragBF a00, a01, a10, a11;
        a00.h[0] = *(const v8bf*)&xb[arow0 * LDSW +      khalf * 8];
        a00.h[1] = *(const v8bf*)&xb[arow0 * LDSW + 16 + khalf * 8];
        a01.h[0] = *(const v8bf*)&xb[arow0 * LDSW + 32 + khalf * 8];
        a01.h[1] = *(const v8bf*)&xb[arow0 * LDSW + 48 + khalf * 8];
        a10.h[0] = *(const v8bf*)&xb[arow1 * LDSW +      khalf * 8];
        a10.h[1] = *(const v8bf*)&xb[arow1 * LDSW + 16 + khalf * 8];
        a11.h[0] = *(const v8bf*)&xb[arow1 * LDSW + 32 + khalf * 8];
        a11.h[1] = *(const v8bf*)&xb[arow1 * LDSW + 48 + khalf * 8];

        // ---- B fragments per N-frag (both K-sections) + 16 WMMA ----
        #pragma unroll
        for (int nf = 0; nf < 4; ++nf) {
            const int brow = wn * 64 + nf * 16 + lrow;
            FragBF b0, b1;
            b0.h[0] = *(const v8bf*)&wb[brow * LDSW +      khalf * 8];
            b0.h[1] = *(const v8bf*)&wb[brow * LDSW + 16 + khalf * 8];
            b1.h[0] = *(const v8bf*)&wb[brow * LDSW + 32 + khalf * 8];
            b1.h[1] = *(const v8bf*)&wb[brow * LDSW + 48 + khalf * 8];
            acc[0][nf] = __builtin_amdgcn_wmma_f32_16x16x32_bf16(
                false, a00.v, false, b0.v, (short)0, acc[0][nf], false, false);
            acc[1][nf] = __builtin_amdgcn_wmma_f32_16x16x32_bf16(
                false, a10.v, false, b0.v, (short)0, acc[1][nf], false, false);
            acc[0][nf] = __builtin_amdgcn_wmma_f32_16x16x32_bf16(
                false, a01.v, false, b1.v, (short)0, acc[0][nf], false, false);
            acc[1][nf] = __builtin_amdgcn_wmma_f32_16x16x32_bf16(
                false, a11.v, false, b1.v, (short)0, acc[1][nf], false, false);
        }
    }

    // ---- epilogue: C layout (VGPR r -> M = r + 8*khalf, N = lane%16) ----
    #pragma unroll
    for (int nf = 0; nf < 4; ++nf) {
        const long  ncol = (long)nblk + wn * 64 + nf * 16 + lrow;
        const float bf   = (float)bias[ncol];
        #pragma unroll
        for (int mf = 0; mf < 2; ++mf) {
            const int mbase = mblk + wm * 32 + mf * 16 + khalf * 8;
            #pragma unroll
            for (int r = 0; r < 8; ++r) {
                out[(long)(mbase + r) * OUT_DIM + ncol] =
                    (__bf16)(acc[mf][nf][r] + bf);
            }
        }
    }
}

extern "C" void kernel_launch(void* const* d_in, const int* in_sizes, int n_in,
                              void* d_out, int out_size, void* d_ws, size_t ws_size,
                              hipStream_t stream)
{
    const __bf16* x      = (const __bf16*)d_in[0];
    const int*    codes  = (const int*)d_in[1];
    const float*  absmax = (const float*)d_in[2];
    const __bf16* bias   = (const __bf16*)d_in[3];
    __bf16*       out    = (__bf16*)d_out;

    dim3 grid(OUT_DIM / TN, S_DIM / TM);   // (86, 16)
    fp4_linear_wmma<<<grid, 256, 0, stream>>>(x, codes, absmax, bias, out);
}
